// FlowGuidedDCN_41085657154038
// MI455X (gfx1250) — matrix operand order repaired and checked by reading, not compiled, not run on hardware
//
#include <hip/hip_runtime.h>

typedef __attribute__((ext_vector_type(16))) _Float16 v16h;
typedef __attribute__((ext_vector_type(8)))  _Float16 v8h;
typedef __attribute__((ext_vector_type(8)))  float    v8f;

#define KK    9
#define CIN   64
#define COUT  64
#define NB    8
#define H_    160
#define W_    160
#define HW_   (H_ * W_)
#define NPIX  (NB * HW_)

// ---- workspace layout (bytes, all offsets 256-aligned) ----
static const size_t WS_INP_H = 0;          // _Float16[NPIX*CIN]  26,214,400 B
static const size_t WS_FEA_H = 26214400;   // _Float16[NPIX*CIN]  26,214,400 B
static const size_t WS_WMAIN = 52428800;   // _Float16[KK*COUT*CIN]   73,728 B
static const size_t WS_WOFF  = 52502528;   // _Float16[KK*32*CIN]     36,864 B
static const size_t WS_PY    = 52539392;   // float[NB*KK*HW]      7,372,800 B
static const size_t WS_PX    = 59912192;   // float[NB*KK*HW]      7,372,800 B
static const size_t WS_MK    = 67284992;   // float[NB*KK*HW]      7,372,800 B

union AFrag {
    v16h v;
    v8h  g[2];
};

static __device__ __forceinline__ v8h splat8(_Float16 x) {
    v8h v = {x, x, x, x, x, x, x, x};
    return v;
}

// ---------------- NCHW f32 -> NHWC f16 ----------------
__global__ void pack_nhwc_f16(const float* __restrict__ src, _Float16* __restrict__ dst) {
    int tid = blockIdx.x * blockDim.x + threadIdx.x;
    if (tid >= NPIX * CIN) return;
    int c   = tid & 63;
    int nhw = tid >> 6;
    int n   = nhw / HW_;
    int hw  = nhw - n * HW_;
    dst[tid] = (_Float16)src[((size_t)n * CIN + c) * HW_ + hw];
}

// ------- pack weights to B-fragment-friendly [k][out][c] f16 -------
__global__ void pack_weights(const float* __restrict__ weight,
                             const float* __restrict__ offw,
                             _Float16* __restrict__ wmain,
                             _Float16* __restrict__ woff) {
    int tid = blockIdx.x * blockDim.x + threadIdx.x;
    const int NM = KK * COUT * CIN;   // 36864
    const int NO = KK * 32 * CIN;     // 18432
    if (tid < NM) {
        int c = tid & 63;
        int o = (tid >> 6) & 63;
        int k = tid >> 12;
        wmain[tid] = (_Float16)weight[((size_t)o * CIN + c) * KK + k];
    } else if (tid < NM + NO) {
        int t = tid - NM;
        int c = t & 63;
        int j = (t >> 6) & 31;
        int k = t >> 11;
        woff[t] = (j < 27) ? (_Float16)offw[((size_t)j * CIN + c) * KK + k] : (_Float16)0.f;
    }
}

// ------- offset conv (implicit GEMM, 16px x 32ch, K=576) + fused offset/mask epilogue -------
__global__ __launch_bounds__(256)
void offset_conv_wmma(const _Float16* __restrict__ fea_h,
                      const _Float16* __restrict__ woff,
                      const float* __restrict__ offb,
                      const float* __restrict__ flows,
                      float* __restrict__ PY, float* __restrict__ PX, float* __restrict__ MK) {
    const int lane = threadIdx.x & 31;
    const int col  = lane & 15;
    const int half = lane >> 4;
    const int wid  = (blockIdx.x * blockDim.x + threadIdx.x) >> 5;
    const int p0   = wid * 16;
    const int p    = p0 + col;
    const int n    = p / HW_;
    const int hw   = p - n * HW_;
    const int h    = hw / W_;
    const int w    = hw - h * W_;

    v8f acc0 = {}; v8f acc1 = {};
    const _Float16* fb = fea_h + (size_t)n * HW_ * CIN;

    for (int k = 0; k < KK; ++k) {
        const int ky = k / 3 - 1, kx = k % 3 - 1;
        const int ys = h + ky, xs = w + kx;
        const bool valid = (ys >= 0) && (ys < H_) && (xs >= 0) && (xs < W_);
        const int ysc = min(max(ys, 0), H_ - 1);
        const int xsc = min(max(xs, 0), W_ - 1);
        const _Float16* bp = fb + ((size_t)ysc * W_ + xsc) * CIN;
        #pragma unroll
        for (int chalf = 0; chalf < 2; ++chalf) {
            AFrag a;
            #pragma unroll
            for (int g = 0; g < 2; ++g) {
                const int cg = chalf * 32 + half * 16 + g * 8;
                v8h v = {};
                if (valid) v = *(const v8h*)(bp + cg);
                a.g[g] = v;
            }
            {
                const v16h b0 = *(const v16h*)(woff + ((size_t)(k * 32 + col) * CIN) + chalf * 32 + half * 16);
                acc0 = __builtin_amdgcn_wmma_f32_16x16x32_f16(false, a.v, false, b0, (short)0, acc0, false, false);
                const v16h b1 = *(const v16h*)(woff + ((size_t)(k * 32 + 16 + col) * CIN) + chalf * 32 + half * 16);
                acc1 = __builtin_amdgcn_wmma_f32_16x16x32_f16(false, a.v, false, b1, (short)0, acc1, false, false);
            }
        }
    }

    // epilogue: channel j -> (py,px,mask) planes with tanh*10 + flipped flow + base grid
    #pragma unroll
    for (int t = 0; t < 2; ++t) {
        #pragma unroll
        for (int r = 0; r < 8; ++r) {
            const int pm  = p0 + r + half * 8;
            const int hwr = pm - n * HW_;
            const int hr  = hwr / W_;
            const int wr  = hwr - hr * W_;
            const int j   = t * 16 + col;
            float v = (t == 0) ? acc0[r] : acc1[r];
            if (j < 27) {
                v += offb[j];
                if (j < 18) {
                    const int kk = j >> 1;
                    // flip: even (dy) channels get flows ch1, odd (dx) get flows ch0
                    const float fl = flows[((size_t)n * 2 + (1 - (j & 1))) * HW_ + hwr];
                    float val = tanhf(v) * 10.f + fl;
                    if ((j & 1) == 0) {
                        val += (float)(hr - 1 + kk / 3);
                        PY[((size_t)n * KK + kk) * HW_ + hwr] = val;
                    } else {
                        val += (float)(wr - 1 + kk % 3);
                        PX[((size_t)n * KK + kk) * HW_ + hwr] = val;
                    }
                } else {
                    const int kk = j - 18;
                    MK[((size_t)n * KK + kk) * HW_ + hwr] = 1.f / (1.f + expf(-v));
                }
            }
        }
    }
}

// ------- deformable implicit GEMM: 16px x 64cout per wave, K = 9*64 -------
__global__ __launch_bounds__(256)
void deform_gemm_wmma(const _Float16* __restrict__ inp_h,
                      const _Float16* __restrict__ wmain,
                      const float* __restrict__ bias,
                      const float* __restrict__ PY, const float* __restrict__ PX,
                      const float* __restrict__ MK,
                      float* __restrict__ out) {
    const int lane = threadIdx.x & 31;
    const int col  = lane & 15;
    const int half = lane >> 4;
    const int wid  = (blockIdx.x * blockDim.x + threadIdx.x) >> 5;
    const int p0   = wid * 16;
    const int p    = p0 + col;
    const int n    = p / HW_;
    const int hw   = p - n * HW_;

    v8f acc[4] = {{}, {}, {}, {}};
    const _Float16* ib = inp_h + (size_t)n * HW_ * CIN;

    for (int k = 0; k < KK; ++k) {
        const size_t oidx = ((size_t)n * KK + k) * HW_ + hw;
        const float py = PY[oidx];
        const float px = PX[oidx];
        const float mk = MK[oidx];
        const float y0f = floorf(py), x0f = floorf(px);
        const int   y0 = (int)y0f,   x0 = (int)x0f;
        const float fy = py - y0f,   fx = px - x0f;
        float wy0 = 1.f - fy, wy1 = fy, wx0 = 1.f - fx, wx1 = fx;
        if (y0 < 0 || y0 >= H_)         wy0 = 0.f;
        if (y0 + 1 < 0 || y0 + 1 >= H_) wy1 = 0.f;
        if (x0 < 0 || x0 >= W_)         wx0 = 0.f;
        if (x0 + 1 < 0 || x0 + 1 >= W_) wx1 = 0.f;
        // fold mask into the 4 corner weights, convert once to f16
        const _Float16 m00 = (_Float16)(wy0 * wx0 * mk);
        const _Float16 m01 = (_Float16)(wy0 * wx1 * mk);
        const _Float16 m10 = (_Float16)(wy1 * wx0 * mk);
        const _Float16 m11 = (_Float16)(wy1 * wx1 * mk);
        const v8h vw00 = splat8(m00), vw01 = splat8(m01);
        const v8h vw10 = splat8(m10), vw11 = splat8(m11);

        const int y0c = min(max(y0, 0), H_ - 1);
        const int y1c = min(max(y0 + 1, 0), H_ - 1);
        const int x0c = min(max(x0, 0), W_ - 1);
        const int x1c = min(max(x0 + 1, 0), W_ - 1);
        const _Float16* p00 = ib + ((size_t)y0c * W_ + x0c) * CIN;
        const _Float16* p01 = ib + ((size_t)y0c * W_ + x1c) * CIN;
        const _Float16* p10 = ib + ((size_t)y1c * W_ + x0c) * CIN;
        const _Float16* p11 = ib + ((size_t)y1c * W_ + x1c) * CIN;

        #pragma unroll
        for (int chalf = 0; chalf < 2; ++chalf) {
            AFrag a;
            #pragma unroll
            for (int g = 0; g < 2; ++g) {
                const int cg = chalf * 32 + half * 16 + g * 8;
                const v8h t00 = *(const v8h*)(p00 + cg);
                const v8h t01 = *(const v8h*)(p01 + cg);
                const v8h t10 = *(const v8h*)(p10 + cg);
                const v8h t11 = *(const v8h*)(p11 + cg);
                // packed f16 bilinear: lowers to v_pk_fma_f16 chains
                v8h s = t00 * vw00;
                s = t01 * vw01 + s;
                s = t10 * vw10 + s;
                s = t11 * vw11 + s;
                a.g[g] = s;
            }
            #pragma unroll
            for (int t = 0; t < 4; ++t) {
                const v16h b = *(const v16h*)(wmain + ((size_t)(k * COUT + t * 16 + col) * CIN)
                                                     + chalf * 32 + half * 16);
                acc[t] = __builtin_amdgcn_wmma_f32_16x16x32_f16(false, a.v, false, b, (short)0, acc[t], false, false);
            }
        }
    }

    #pragma unroll
    for (int t = 0; t < 4; ++t) {
        const int o = t * 16 + col;
        const float bi = bias[o];
        float* ob = out + ((size_t)n * COUT + o) * HW_;
        #pragma unroll
        for (int r = 0; r < 8; ++r) {
            const int pm  = p0 + r + half * 8;
            const int hwr = pm - n * HW_;
            ob[hwr] = acc[t][r] + bi;
        }
    }
}

extern "C" void kernel_launch(void* const* d_in, const int* in_sizes, int n_in,
                              void* d_out, int out_size, void* d_ws, size_t ws_size,
                              hipStream_t stream) {
    const float* input  = (const float*)d_in[0];
    const float* fea    = (const float*)d_in[1];
    const float* flows  = (const float*)d_in[2];
    const float* offw   = (const float*)d_in[3];
    const float* offb   = (const float*)d_in[4];
    const float* weight = (const float*)d_in[5];
    const float* bias   = (const float*)d_in[6];
    float* out = (float*)d_out;
    char*  ws  = (char*)d_ws;

    _Float16* inp_h = (_Float16*)(ws + WS_INP_H);
    _Float16* fea_h = (_Float16*)(ws + WS_FEA_H);
    _Float16* wmain = (_Float16*)(ws + WS_WMAIN);
    _Float16* woff  = (_Float16*)(ws + WS_WOFF);
    float* PY = (float*)(ws + WS_PY);
    float* PX = (float*)(ws + WS_PX);
    float* MK = (float*)(ws + WS_MK);

    const int packN = NPIX * CIN;
    pack_nhwc_f16<<<(packN + 255) / 256, 256, 0, stream>>>(input, inp_h);
    pack_nhwc_f16<<<(packN + 255) / 256, 256, 0, stream>>>(fea, fea_h);
    pack_weights<<<(36864 + 18432 + 255) / 256, 256, 0, stream>>>(weight, offw, wmain, woff);

    const int ntiles  = NPIX / 16;   // 12800 wave-tiles
    const int nblocks = ntiles / 8;  // 8 waves per 256-thread block
    offset_conv_wmma<<<nblocks, 256, 0, stream>>>(fea_h, woff, offb, flows, PY, PX, MK);
    deform_gemm_wmma<<<nblocks, 256, 0, stream>>>(inp_h, wmain, bias, PY, PX, MK, out);
}
